// DoorLocalModel_57432302682461
// MI455X (gfx1250) — compile-verified, matrix operand order checked by reading, not run on hardware
//
#include <hip/hip_runtime.h>
#include <hip/hip_bf16.h>
#include <stdint.h>

// ---------------------------------------------------------------------------
// DoorLocalModel forward on gfx1250 (CDNA5), f16 WMMA with f32 accumulation.
// ---------------------------------------------------------------------------

typedef __attribute__((ext_vector_type(16))) _Float16 v16h;
typedef __attribute__((ext_vector_type(8)))  _Float16 v8h;
typedef __attribute__((ext_vector_type(8)))  float    v8f;

#define NB     512      // batch (envs)
#define DPT    4096     // doors per table
#define NDOOR  16384    // 4 tables
#define LW_    256
#define GW_    512
#define ND_    600
#define KD     352      // door GEMM K (324 padded to x32)
#define KG     320      // global GEMM K (303 padded)
#define KL     768      // local GEMM K
#define KF     512      // fc / sv GEMM K
#define SVCOLS 608      // sv columns computed (>= 600, x16)

__device__ __forceinline__ int imin(int a, int b) { return a < b ? a : b; }
__device__ __forceinline__ int imax(int a, int b) { return a > b ? a : b; }

// Load one 16x32 f16 fragment (A or B role) from a row-major buffer.
// Per CDNA5 ISA 16-bit matrix striping: lanes 0-15 hold row=lane,
// halves {K0..7, K16..23}; lanes 16-31 hold row=lane-16, halves
// {K8..15, K24..31}.  Row-major storage makes each group 8 contiguous
// halves -> two 16-byte loads.
__device__ __forceinline__ v16h frag_pair(const _Float16* p) {
  v16h r;
  *(v8h*)&r         = *(const v8h*)p;
  *(((v8h*)&r) + 1) = *(const v8h*)(p + 16);
  return r;
}
__device__ __forceinline__ v16h load_frag(const _Float16* base, int ld) {
  const int lane = threadIdx.x & 31;
  return frag_pair(base + (size_t)(lane & 15) * ld + ((lane & 16) ? 8 : 0));
}

__device__ __forceinline__ v8f wmma_f16(v16h a, v16h b, v8f c) {
  return __builtin_amdgcn_wmma_f32_16x16x32_f16(
      /*neg_a=*/false, a, /*neg_b=*/false, b,
      /*c_mod=*/(short)0, c, /*reuse_a=*/false, /*reuse_b=*/false);
}

// ---------------------------------------------------------------------------
// Weight / input conversion helpers
// ---------------------------------------------------------------------------
__global__ void k_cvt(const float* __restrict__ src, _Float16* __restrict__ dst,
                      int rows, int scols, int dcols, int sstride, int soffs) {
  int n = rows * dcols;
  for (int i = blockIdx.x * blockDim.x + threadIdx.x; i < n;
       i += gridDim.x * blockDim.x) {
    int r = i / dcols, c = i - r * dcols;
    float v = (c < scols) ? src[(size_t)r * sstride + soffs + c] : 0.f;
    dst[i] = (_Float16)v;
  }
}

// gin (f16, NB x KG): [room_mask(300) | steps/100 | round_frac | log(temp) | pad]
__global__ void k_gin(const float* __restrict__ rm, const float* __restrict__ steps,
                      const float* __restrict__ rf, const float* __restrict__ temp,
                      _Float16* __restrict__ gin) {
  int n = NB * KG;
  for (int i = blockIdx.x * blockDim.x + threadIdx.x; i < n;
       i += gridDim.x * blockDim.x) {
    int e = i / KG, k = i - e * KG;
    float v = 0.f;
    if (k < 300)       v = rm[e * 300 + k];
    else if (k == 300) v = steps[e] * 0.01f;
    else if (k == 301) v = rf[e];
    else if (k == 302) v = logf(temp[e]);
    gin[i] = (_Float16)v;
  }
}

__global__ void k_copy(const float* __restrict__ src, float* __restrict__ dst, int n) {
  for (int i = blockIdx.x * blockDim.x + threadIdx.x; i < n;
       i += gridDim.x * blockDim.x)
    dst[i] = src[i];
}

// ---------------------------------------------------------------------------
// Door patch GEMM: C(16 x 512) = patches(16 x 324) @ W^T, fused
// bias + maxout(j, j+256) + pe_x/pe_y/door_emb add.  1 block = 16 doors.
// ---------------------------------------------------------------------------
__global__ __launch_bounds__(256) void k_door_gemm(
    const float* __restrict__ map, const int* __restrict__ tbl,
    const _Float16* __restrict__ wh, const float* __restrict__ bias,
    const float* __restrict__ pe_x, const float* __restrict__ pe_y,
    const float* __restrict__ demb, float* __restrict__ localX, int row_base) {
  __shared__ __align__(16) _Float16 sA[16 * KD];
  __shared__ int sMeta[64];
  const int tid = threadIdx.x;
  const int m0  = blockIdx.x * 16;
  if (tid < 64) sMeta[tid] = tbl[m0 * 4 + tid];
  __syncthreads();
  // build f16 patch tile: layout k = kx*36 + ky*4 + c  (matches (K,K,C) reshape)
  for (int idx = tid; idx < 16 * KD; idx += 256) {
    int d = idx / KD, kk = idx - d * KD;
    float v = 0.f;
    if (kk < 324) {
      int e  = sMeta[d * 4 + 0];
      int px = sMeta[d * 4 + 2] - 4;
      int py = sMeta[d * 4 + 3] - 4;
      int kx = kk / 36, r = kk - kx * 36;
      int ky = r >> 2, c = r & 3;
      int x = imin(imax(px + kx, 0), 72);
      int y = imin(imax(py + ky, 0), 72);
      v = map[((size_t)(e * 4 + c) * 73 + x) * 73 + y];
    }
    sA[idx] = (_Float16)v;
  }
  __syncthreads();
  const int wave = tid >> 5, lane = tid & 31;
  v8f acc[4] = {};
  for (int k0 = 0; k0 < KD; k0 += 32) {
    v16h a = load_frag(sA + k0, KD);
#pragma unroll
    for (int p = 0; p < 2; ++p) {
      int n0 = (wave * 2 + p) * 16;
      v16h b0 = load_frag(wh + (size_t)n0 * KD + k0, KD);
      acc[2 * p] = wmma_f16(a, b0, acc[2 * p]);
      v16h b1 = load_frag(wh + (size_t)(n0 + 256) * KD + k0, KD);
      acc[2 * p + 1] = wmma_f16(a, b1, acc[2 * p + 1]);
    }
  }
  const int nl = lane & 15, mo = (lane & 16) ? 8 : 0;
#pragma unroll
  for (int p = 0; p < 2; ++p) {
    int n = (wave * 2 + p) * 16 + nl;
    float b0 = bias[n], b1 = bias[n + 256];
#pragma unroll
    for (int r = 0; r < 8; ++r) {
      int m = r + mo;
      float v = fmaxf(acc[2 * p][r] + b0, acc[2 * p + 1][r] + b1);
      int did = sMeta[m * 4 + 1], px = sMeta[m * 4 + 2], py = sMeta[m * 4 + 3];
      v += pe_x[px * LW_ + n] + pe_y[py * LW_ + n] + demb[did * LW_ + n];
      localX[(size_t)(row_base + m0 + m) * LW_ + n] = v;
    }
  }
}

// ---------------------------------------------------------------------------
// global_X = maxout(gin @ w_global_sub^T + b).  1 block = 16 envs x 512 cols.
// ---------------------------------------------------------------------------
__global__ __launch_bounds__(256) void k_global_gemm(
    const _Float16* __restrict__ gin, const _Float16* __restrict__ wh,
    const float* __restrict__ bias, float* __restrict__ gX) {
  const int tid = threadIdx.x, wave = tid >> 5, lane = tid & 31;
  const int m0 = blockIdx.x * 16;
  v8f acc[8] = {};
  for (int k0 = 0; k0 < KG; k0 += 32) {
    v16h a = load_frag(gin + (size_t)m0 * KG + k0, KG);
#pragma unroll
    for (int p = 0; p < 4; ++p) {
      int n0 = (wave * 4 + p) * 16;
      v16h b0 = load_frag(wh + (size_t)n0 * KG + k0, KG);
      acc[2 * p] = wmma_f16(a, b0, acc[2 * p]);
      v16h b1 = load_frag(wh + (size_t)(n0 + 512) * KG + k0, KG);
      acc[2 * p + 1] = wmma_f16(a, b1, acc[2 * p + 1]);
    }
  }
  const int nl = lane & 15, mo = (lane & 16) ? 8 : 0;
#pragma unroll
  for (int p = 0; p < 4; ++p) {
    int n = (wave * 4 + p) * 16 + nl;
    float b0 = bias[n], b1 = bias[n + 512];
#pragma unroll
    for (int r = 0; r < 8; ++r) {
      int m = m0 + r + mo;
      gX[(size_t)m * GW_ + n] = fmaxf(acc[2 * p][r] + b0, acc[2 * p + 1][r] + b1);
    }
  }
}

// ---------------------------------------------------------------------------
// Local round: comb = maxout([local | global[env]] @ w_loc^T + b)
//   cols <256  -> lxOut = lxIn + comb
//   cols >=256 -> atomic segment-sum into gxOut[env]
// grid (1024, 3): x = 16-row tile, y = 256-maxout-col group.
// ---------------------------------------------------------------------------
__global__ __launch_bounds__(256) void k_loc_gemm(
    const float* __restrict__ lxIn, const float* __restrict__ gxIn,
    const _Float16* __restrict__ wh, const float* __restrict__ bias,
    const int* __restrict__ t0, const int* __restrict__ t1,
    const int* __restrict__ t2, const int* __restrict__ t3,
    float* __restrict__ lxOut, float* __restrict__ gxOut) {
  __shared__ __align__(16) _Float16 sA[16 * KL];
  __shared__ int sEnv[16];
  const int tid = threadIdx.x;
  const int m0  = blockIdx.x * 16;
  if (tid < 16) {
    int g = m0 + tid;
    const int* tb = (g < 4096) ? t0 : (g < 8192) ? t1 : (g < 12288) ? t2 : t3;
    sEnv[tid] = tb[(g & 4095) * 4];
  }
  __syncthreads();
  for (int idx = tid; idx < 16 * KL; idx += 256) {
    int d = idx / KL, k = idx - d * KL;
    float v = (k < LW_) ? lxIn[(size_t)(m0 + d) * LW_ + k]
                        : gxIn[(size_t)sEnv[d] * GW_ + (k - LW_)];
    sA[idx] = (_Float16)v;
  }
  __syncthreads();
  const int wave = tid >> 5, lane = tid & 31;
  const int nbase = blockIdx.y * 256;
  v8f acc[4] = {};
  for (int k0 = 0; k0 < KL; k0 += 32) {
    v16h a = load_frag(sA + k0, KL);
#pragma unroll
    for (int p = 0; p < 2; ++p) {
      int n0 = nbase + (wave * 2 + p) * 16;
      v16h b0 = load_frag(wh + (size_t)n0 * KL + k0, KL);
      acc[2 * p] = wmma_f16(a, b0, acc[2 * p]);
      v16h b1 = load_frag(wh + (size_t)(n0 + KL) * KL + k0, KL);
      acc[2 * p + 1] = wmma_f16(a, b1, acc[2 * p + 1]);
    }
  }
  const int nl = lane & 15, mo = (lane & 16) ? 8 : 0;
#pragma unroll
  for (int p = 0; p < 2; ++p) {
    int n = nbase + (wave * 2 + p) * 16 + nl;
    float b0 = bias[n], b1 = bias[n + KL];
#pragma unroll
    for (int r = 0; r < 8; ++r) {
      int m = r + mo, g = m0 + m;
      float v = fmaxf(acc[2 * p][r] + b0, acc[2 * p + 1][r] + b1);
      if (n < LW_) {
        lxOut[(size_t)g * LW_ + n] = lxIn[(size_t)g * LW_ + n] + v;
      } else {
        atomicAdd(&gxOut[(size_t)sEnv[m] * GW_ + (n - LW_)], v);
      }
    }
  }
}

// ---------------------------------------------------------------------------
// fc round: gxOut = gxIn + maxout(gxIn @ w^T + b).  1 block = 16 envs.
// ---------------------------------------------------------------------------
__global__ __launch_bounds__(256) void k_fc_gemm(
    const float* __restrict__ gxIn, const _Float16* __restrict__ wh,
    const float* __restrict__ bias, float* __restrict__ gxOut) {
  __shared__ __align__(16) _Float16 sA[16 * KF];
  const int tid = threadIdx.x, m0 = blockIdx.x * 16;
  for (int idx = tid; idx < 16 * KF; idx += 256) {
    int d = idx / KF, k = idx - d * KF;
    sA[idx] = (_Float16)gxIn[(size_t)(m0 + d) * GW_ + k];
  }
  __syncthreads();
  const int wave = tid >> 5, lane = tid & 31;
  v8f acc[8] = {};
  for (int k0 = 0; k0 < KF; k0 += 32) {
    v16h a = load_frag(sA + k0, KF);
#pragma unroll
    for (int p = 0; p < 4; ++p) {
      int n0 = (wave * 4 + p) * 16;
      v16h b0 = load_frag(wh + (size_t)n0 * KF + k0, KF);
      acc[2 * p] = wmma_f16(a, b0, acc[2 * p]);
      v16h b1 = load_frag(wh + (size_t)(n0 + 512) * KF + k0, KF);
      acc[2 * p + 1] = wmma_f16(a, b1, acc[2 * p + 1]);
    }
  }
  const int nl = lane & 15, mo = (lane & 16) ? 8 : 0;
#pragma unroll
  for (int p = 0; p < 4; ++p) {
    int n = (wave * 4 + p) * 16 + nl;
    float b0 = bias[n], b1 = bias[n + 512];
#pragma unroll
    for (int r = 0; r < 8; ++r) {
      int m = m0 + r + mo;
      gxOut[(size_t)m * GW_ + n] =
          gxIn[(size_t)m * GW_ + n] + fmaxf(acc[2 * p][r] + b0, acc[2 * p + 1][r] + b1);
    }
  }
}

// ---------------------------------------------------------------------------
// sv = global_X @ w_sv^T + b_sv (first 608 cols).  grid (32, 5).
// ---------------------------------------------------------------------------
__global__ __launch_bounds__(256) void k_sv_gemm(
    const float* __restrict__ gx, const _Float16* __restrict__ wh,
    const float* __restrict__ bias, float* __restrict__ sv) {
  __shared__ __align__(16) _Float16 sA[16 * KF];
  const int tid = threadIdx.x, m0 = blockIdx.x * 16;
  for (int idx = tid; idx < 16 * KF; idx += 256) {
    int d = idx / KF, k = idx - d * KF;
    sA[idx] = (_Float16)gx[(size_t)(m0 + d) * GW_ + k];
  }
  __syncthreads();
  const int wave = tid >> 5, lane = tid & 31;
  const int nt = blockIdx.y * 8 + wave;
  if (nt >= SVCOLS / 16) return;     // uniform per wave: EXEC stays all-1s
  const int n0 = nt * 16;
  v8f acc = {};
  for (int k0 = 0; k0 < KF; k0 += 32) {
    v16h a = load_frag(sA + k0, KF);
    v16h b = load_frag(wh + (size_t)n0 * KF + k0, KF);
    acc = wmma_f16(a, b, acc);
  }
  const int nl = lane & 15, mo = (lane & 16) ? 8 : 0;
  int n = n0 + nl;
  float bb = bias[n];
#pragma unroll
  for (int r = 0; r < 8; ++r) {
    int m = m0 + r + mo;
    sv[(size_t)m * SVCOLS + n] = acc[r] + bb;
  }
}

// ---------------------------------------------------------------------------
// door_logodds[row] = dot(local_X[row], w_doorlog) + b.  1 wave / row.
// ---------------------------------------------------------------------------
__global__ __launch_bounds__(256) void k_doorlog(
    const float* __restrict__ lx, const float* __restrict__ w,
    const float* __restrict__ b, float* __restrict__ dl) {
  const int wave = threadIdx.x >> 5, lane = threadIdx.x & 31;
  const int row = blockIdx.x * 8 + wave;
  const float* p = lx + (size_t)row * LW_;
  float s = 0.f;
#pragma unroll
  for (int i = 0; i < 8; ++i) {
    int k = lane * 8 + i;
    s += p[k] * w[k];
  }
#pragma unroll
  for (int off = 16; off >= 1; off >>= 1) s += __shfl_xor(s, off, 32);
  if (lane == 0) dl[row] = s + b[0];
}

// ---------------------------------------------------------------------------
// Output assembly.
// ---------------------------------------------------------------------------
__global__ void k_out_init(const float* __restrict__ sv,
                           const unsigned char* __restrict__ dc,
                           float* __restrict__ out) {
  int n = NB * ND_;
  for (int i = blockIdx.x * blockDim.x + threadIdx.x; i < n;
       i += gridDim.x * blockDim.x) {
    int e = i / ND_, d = i - e * ND_;
    out[i] = dc[i] ? 100000.0f : sv[(size_t)e * SVCOLS + d];
  }
}

__global__ void k_out_scatter(const int* __restrict__ t0, const int* __restrict__ t1,
                              const int* __restrict__ t2, const int* __restrict__ t3,
                              const float* __restrict__ dl,
                              const unsigned char* __restrict__ dc,
                              float* __restrict__ out) {
  int g = blockIdx.x * blockDim.x + threadIdx.x;
  if (g >= NDOOR) return;
  const int* tb = (g < 4096) ? t0 : (g < 8192) ? t1 : (g < 12288) ? t2 : t3;
  int i = (g & 4095) * 4;
  int p = tb[i] * ND_ + tb[i + 1];
  if (!dc[p]) out[p] = dl[g];     // positions are unique (permutation) -> no race
}

// ---------------------------------------------------------------------------
extern "C" void kernel_launch(void* const* d_in, const int* in_sizes, int n_in,
                              void* d_out, int out_size, void* d_ws, size_t ws_size,
                              hipStream_t stream) {
  (void)in_sizes; (void)n_in; (void)out_size; (void)ws_size;
  const float* map    = (const float*)d_in[0];
  const float* rm     = (const float*)d_in[1];
  const float* steps  = (const float*)d_in[2];
  const float* rfrac  = (const float*)d_in[3];
  const float* temp   = (const float*)d_in[4];
  const float* pe_x   = (const float*)d_in[5];
  const float* pe_y   = (const float*)d_in[6];
  const float* demb   = (const float*)d_in[7];
  const float* w_left = (const float*)d_in[8];   const float* b_left = (const float*)d_in[9];
  const float* w_right= (const float*)d_in[10];  const float* b_right= (const float*)d_in[11];
  const float* w_up   = (const float*)d_in[12];  const float* b_up   = (const float*)d_in[13];
  const float* w_down = (const float*)d_in[14];  const float* b_down = (const float*)d_in[15];
  const float* w_glob = (const float*)d_in[16];  const float* b_glob = (const float*)d_in[17];
  const float* w_loc0 = (const float*)d_in[18];  const float* b_loc0 = (const float*)d_in[19];
  const float* w_loc1 = (const float*)d_in[20];  const float* b_loc1 = (const float*)d_in[21];
  const float* w_dlog = (const float*)d_in[22];  const float* b_dlog = (const float*)d_in[23];
  const float* w_fc0  = (const float*)d_in[24];  const float* b_fc0  = (const float*)d_in[25];
  const float* w_fc1  = (const float*)d_in[26];  const float* b_fc1  = (const float*)d_in[27];
  const float* w_sv   = (const float*)d_in[28];  const float* b_sv   = (const float*)d_in[29];
  const int* t_left   = (const int*)d_in[30];
  const int* t_right  = (const int*)d_in[31];
  const int* t_down   = (const int*)d_in[32];
  const int* t_up     = (const int*)d_in[33];
  const unsigned char* dconn = (const unsigned char*)d_in[34];
  float* out = (float*)d_out;

  char* ws = (char*)d_ws;
  size_t off = 0;
  auto alloc = [&](size_t bytes) -> void* {
    void* p = ws + off;
    off = (off + bytes + 255) & ~(size_t)255;
    return p;
  };
  _Float16* hwd  = (_Float16*)alloc((size_t)4 * 512 * KD * 2);
  _Float16* hwl0 = (_Float16*)alloc((size_t)1536 * KL * 2);
  _Float16* hwl1 = (_Float16*)alloc((size_t)1536 * KL * 2);
  _Float16* hwg  = (_Float16*)alloc((size_t)1024 * KG * 2);
  _Float16* hwf0 = (_Float16*)alloc((size_t)1024 * KF * 2);
  _Float16* hwf1 = (_Float16*)alloc((size_t)1024 * KF * 2);
  _Float16* hwsv = (_Float16*)alloc((size_t)SVCOLS * KF * 2);
  _Float16* hgin = (_Float16*)alloc((size_t)NB * KG * 2);
  float* lx0 = (float*)alloc((size_t)NDOOR * LW_ * 4);
  float* lx1 = (float*)alloc((size_t)NDOOR * LW_ * 4);
  float* gx0 = (float*)alloc((size_t)NB * GW_ * 4);
  float* gx1 = (float*)alloc((size_t)NB * GW_ * 4);
  float* dl  = (float*)alloc((size_t)NDOOR * 4);
  float* svb = (float*)alloc((size_t)NB * SVCOLS * 4);

  dim3 B(256);
  // --- f16 weight conversion (K padded with zeros) ---
  k_cvt<<<256, B, 0, stream>>>(w_left,  hwd + (size_t)0 * 512 * KD, 512, 324, KD, 324, 0);
  k_cvt<<<256, B, 0, stream>>>(w_right, hwd + (size_t)1 * 512 * KD, 512, 324, KD, 324, 0);
  k_cvt<<<256, B, 0, stream>>>(w_down,  hwd + (size_t)2 * 512 * KD, 512, 324, KD, 324, 0);
  k_cvt<<<256, B, 0, stream>>>(w_up,    hwd + (size_t)3 * 512 * KD, 512, 324, KD, 324, 0);
  k_cvt<<<512, B, 0, stream>>>(w_loc0, hwl0, 1536, KL, KL, KL, 0);
  k_cvt<<<512, B, 0, stream>>>(w_loc1, hwl1, 1536, KL, KL, KL, 0);
  k_cvt<<<256, B, 0, stream>>>(w_glob, hwg, 1024, 303, KG, 4399, 4096);  // rc_flat cols are 0
  k_cvt<<<256, B, 0, stream>>>(w_fc0, hwf0, 1024, KF, KF, KF, 0);
  k_cvt<<<256, B, 0, stream>>>(w_fc1, hwf1, 1024, KF, KF, KF, 0);
  k_cvt<<<256, B, 0, stream>>>(w_sv,  hwsv, SVCOLS, KF, KF, KF, 0);
  k_gin<<<128, B, 0, stream>>>(rm, steps, rfrac, temp, hgin);

  // --- 4 door GEMMs (concat order: left, right, down, up) ---
  k_door_gemm<<<256, B, 0, stream>>>(map, t_left,  hwd + (size_t)0 * 512 * KD, b_left,  pe_x, pe_y, demb, lx0, 0);
  k_door_gemm<<<256, B, 0, stream>>>(map, t_right, hwd + (size_t)1 * 512 * KD, b_right, pe_x, pe_y, demb, lx0, 4096);
  k_door_gemm<<<256, B, 0, stream>>>(map, t_down,  hwd + (size_t)2 * 512 * KD, b_down,  pe_x, pe_y, demb, lx0, 8192);
  k_door_gemm<<<256, B, 0, stream>>>(map, t_up,    hwd + (size_t)3 * 512 * KD, b_up,    pe_x, pe_y, demb, lx0, 12288);

  // --- global input GEMM ---
  k_global_gemm<<<32, B, 0, stream>>>(hgin, hwg, b_glob, gx0);

  // --- local/global interaction rounds (ping-pong) ---
  k_copy<<<256, B, 0, stream>>>(gx0, gx1, NB * GW_);
  k_loc_gemm<<<dim3(1024, 3), B, 0, stream>>>(lx0, gx0, hwl0, b_loc0,
                                              t_left, t_right, t_down, t_up, lx1, gx1);
  k_copy<<<256, B, 0, stream>>>(gx1, gx0, NB * GW_);
  k_loc_gemm<<<dim3(1024, 3), B, 0, stream>>>(lx1, gx1, hwl1, b_loc1,
                                              t_left, t_right, t_down, t_up, lx0, gx0);

  // --- door logit head ---
  k_doorlog<<<2048, B, 0, stream>>>(lx0, w_dlog, b_dlog, dl);

  // --- global fc rounds ---
  k_fc_gemm<<<32, B, 0, stream>>>(gx0, hwf0, b_fc0, gx1);
  k_fc_gemm<<<32, B, 0, stream>>>(gx1, hwf1, b_fc1, gx0);

  // --- state-value head ---
  k_sv_gemm<<<dim3(32, 5), B, 0, stream>>>(gx0, hwsv, b_sv, svb);

  // --- output assembly ---
  k_out_init<<<512, B, 0, stream>>>(svb, dconn, out);
  k_out_scatter<<<64, B, 0, stream>>>(t_left, t_right, t_down, t_up, dl, dconn, out);
}
